// GRUCell_11879879541673
// MI455X (gfx1250) — matrix operand order repaired and checked by reading, not compiled, not run on hardware
//
#include <hip/hip_runtime.h>
#include <cstdint>

#define BB     4096
#define KD     2048
#define NU     2048
#define MSZ    ((size_t)KD * NU)   // elements in one 2048x2048 matrix

typedef __attribute__((ext_vector_type(16))) __bf16 v16bf;
typedef __attribute__((ext_vector_type(8)))  float  v8f;

union FragU {
    uint4  q[2];
    v16bf  v;
};

// ---------------- helpers ----------------

static __device__ __forceinline__ unsigned short f2bf(float f) {
    union { float f; unsigned u; } v; v.f = f;
    unsigned r = v.u + 0x7fffu + ((v.u >> 16) & 1u);   // round-to-nearest-even
    return (unsigned short)(r >> 16);
}

// A operand: 16x32 bf16 tile, row-major source [rows][ld] (bf16 bits in ushort).
// lanes 0-15: M=lane, halfs 0..7 = K k+0..7,  halfs 8..15 = K k+16..23
// lanes16-31: M=lane-16, halfs 0..7 = K k+8..15, halfs 8..15 = K k+24..31
static __device__ __forceinline__ v16bf load_frag_a(const unsigned short* __restrict__ base,
                                                    int ld, int row0, int k, int lane) {
    int m  = lane & 15;
    int hi = lane >> 4;
    const unsigned short* p = base + (size_t)(row0 + m) * ld + k + hi * 8;
    FragU f;
    f.q[0] = *(const uint4*)(p);
    f.q[1] = *(const uint4*)(p + 16);
    return f.v;
}

// B operand: 32x16 bf16 tile from W^T stored [N][K] row-major.
// lanes 0-15: N=lane, K = k+0..7,k+16..23 ; lanes 16-31: N=lane-16, K = k+8..15,k+24..31
static __device__ __forceinline__ v16bf load_frag_b(const unsigned short* __restrict__ base,
                                                    int ld, int col0, int k, int lane) {
    int n  = lane & 15;
    int hi = lane >> 4;
    const unsigned short* p = base + (size_t)(col0 + n) * ld + k + hi * 8;
    FragU f;
    f.q[0] = *(const uint4*)(p);
    f.q[1] = *(const uint4*)(p + 16);
    return f.v;
}

static __device__ __forceinline__ v8f wmma_bf16(v16bf a, v16bf b, v8f c) {
    return __builtin_amdgcn_wmma_f32_16x16x32_bf16(false, a, false, b, (short)0, c, false, false);
}

static __device__ __forceinline__ float sigmoidf_dev(float x) {
    return 1.0f / (1.0f + __expf(-x));
}

// ---------------- prep kernels ----------------

__global__ __launch_bounds__(256) void cvt_f32_bf16(const float* __restrict__ src,
                                                    unsigned short* __restrict__ dst,
                                                    int n4 /* count/4 */) {
    int i = blockIdx.x * blockDim.x + threadIdx.x;
    if (i >= n4) return;
    float4 f = ((const float4*)src)[i];
    ushort4 o;
    o.x = f2bf(f.x); o.y = f2bf(f.y); o.z = f2bf(f.z); o.w = f2bf(f.w);
    ((ushort4*)dst)[i] = o;
}

// dst[n*KD + k] = bf16(src[k*NU + n]); 2048x2048 sub-matrix transpose+convert
__global__ __launch_bounds__(256) void transpose_cvt(const float* __restrict__ src,
                                                     unsigned short* __restrict__ dst) {
    __shared__ float t[32][33];
    int tx = threadIdx.x & 31;
    int ty = threadIdx.x >> 5;            // 0..7
    int k0 = blockIdx.y * 32;
    int n0 = blockIdx.x * 32;
#pragma unroll
    for (int j = 0; j < 32; j += 8)
        t[ty + j][tx] = src[(size_t)(k0 + ty + j) * NU + n0 + tx];
    __syncthreads();
#pragma unroll
    for (int j = 0; j < 32; j += 8)
        dst[(size_t)(n0 + ty + j) * KD + k0 + tx] = f2bf(t[tx][ty + j]);
}

// ---------------- fused gates GEMM: r, z, Xn ----------------
// per wave: 32x16 output tile (two 16x16 M sub-tiles per gate); every B frag
// feeds two WMMAs. Block = 8 waves (2 M x 4 N) = 64x64 tile. Explicit
// double-buffered fragment sets to overlap global loads with WMMAs.

struct GateFrags {
    v16bf ax0, ax1, ah0, ah1;             // A: X rows [r0,r0+16), [r0+16,r0+32); same for H
    v16bf brx, brh, bzx, bzh, bnx;        // B: one 32x16 frag per weight
};

static __device__ __forceinline__ void load_gate_frags(
        GateFrags& f,
        const unsigned short* __restrict__ Xbf, const unsigned short* __restrict__ Hbf,
        const unsigned short* __restrict__ Wrx, const unsigned short* __restrict__ Wrh,
        const unsigned short* __restrict__ Wzx, const unsigned short* __restrict__ Wzh,
        const unsigned short* __restrict__ Wnx,
        int row0, int col0, int k, int lane) {
    f.ax0 = load_frag_a(Xbf, KD, row0,      k, lane);
    f.ax1 = load_frag_a(Xbf, KD, row0 + 16, k, lane);
    f.ah0 = load_frag_a(Hbf, KD, row0,      k, lane);
    f.ah1 = load_frag_a(Hbf, KD, row0 + 16, k, lane);
    f.brx = load_frag_b(Wrx, KD, col0, k, lane);
    f.brh = load_frag_b(Wrh, KD, col0, k, lane);
    f.bzx = load_frag_b(Wzx, KD, col0, k, lane);
    f.bzh = load_frag_b(Wzh, KD, col0, k, lane);
    f.bnx = load_frag_b(Wnx, KD, col0, k, lane);
}

static __device__ __forceinline__ void gate_mma(const GateFrags& f,
        v8f& r0, v8f& r1, v8f& z0, v8f& z1, v8f& n0, v8f& n1) {
    r0 = wmma_bf16(f.ax0, f.brx, r0);
    r1 = wmma_bf16(f.ax1, f.brx, r1);
    r0 = wmma_bf16(f.ah0, f.brh, r0);
    r1 = wmma_bf16(f.ah1, f.brh, r1);
    z0 = wmma_bf16(f.ax0, f.bzx, z0);
    z1 = wmma_bf16(f.ax1, f.bzx, z1);
    z0 = wmma_bf16(f.ah0, f.bzh, z0);
    z1 = wmma_bf16(f.ah1, f.bzh, z1);
    n0 = wmma_bf16(f.ax0, f.bnx, n0);
    n1 = wmma_bf16(f.ax1, f.bnx, n1);
}

__global__ __launch_bounds__(256) void gru_gates_gemm(
        const unsigned short* __restrict__ Xbf,
        const unsigned short* __restrict__ Hbf,
        const unsigned short* __restrict__ Wt,     // 6 matrices [N][K] bf16
        const float* __restrict__ states,
        const float* __restrict__ b_r,
        const float* __restrict__ b_z,
        float* __restrict__ Zout,
        float* __restrict__ NXout,
        unsigned short* __restrict__ HRbf) {
    const int lane = threadIdx.x & 31;
    const int wave = threadIdx.x >> 5;
    const int wm = wave & 1;              // 0..1
    const int wn = wave >> 1;             // 0..3
    const int row0 = blockIdx.y * 64 + wm * 32;
    const int col0 = blockIdx.x * 64 + wn * 16;

    const unsigned short* Wrx = Wt + 0 * MSZ;
    const unsigned short* Wrh = Wt + 1 * MSZ;
    const unsigned short* Wzx = Wt + 2 * MSZ;
    const unsigned short* Wzh = Wt + 3 * MSZ;
    const unsigned short* Wnx = Wt + 4 * MSZ;

    v8f r0 = {}, r1 = {}, z0 = {}, z1 = {}, n0a = {}, n1a = {};

    GateFrags f0, f1;
    load_gate_frags(f0, Xbf, Hbf, Wrx, Wrh, Wzx, Wzh, Wnx, row0, col0, 0, lane);

    for (int k = 0; k < KD; k += 64) {
        load_gate_frags(f1, Xbf, Hbf, Wrx, Wrh, Wzx, Wzh, Wnx, row0, col0, k + 32, lane);
        gate_mma(f0, r0, r1, z0, z1, n0a, n1a);
        if (k + 64 < KD)
            load_gate_frags(f0, Xbf, Hbf, Wrx, Wrh, Wzx, Wzh, Wnx, row0, col0, k + 64, lane);
        gate_mma(f1, r0, r1, z0, z1, n0a, n1a);
    }

    // epilogue: C/D layout — N = lane&15, M = vgpr + 8*(lane>=16)
    const int n0 = lane & 15;
    const int hi = lane >> 4;
    const int col = col0 + n0;
    const float br = b_r[col];
    const float bz = b_z[col];
#pragma unroll
    for (int s = 0; s < 2; ++s) {
        const v8f& accr = s ? r1 : r0;
        const v8f& accz = s ? z1 : z0;
        const v8f& accn = s ? n1a : n0a;
#pragma unroll
        for (int v = 0; v < 8; ++v) {
            const int row = row0 + s * 16 + hi * 8 + v;
            const size_t idx = (size_t)row * NU + col;
            float r = sigmoidf_dev(accr[v] + br);
            float z = sigmoidf_dev(accz[v] + bz);
            Zout[idx]  = z;
            NXout[idx] = accn[v];
            HRbf[idx]  = f2bf(states[idx] * r);
        }
    }
}

// ---------------- final GEMM: (h*r) @ Wn_h, tanh, blend ----------------
// per wave: 32x32 output tile (2x2 of 16x16); each A/B frag feeds 2 WMMAs.
// Block = 8 waves (2 M x 4 N) = 64x128. Double-buffered fragments.

struct NFrags { v16bf a0, a1, b0, b1; };

static __device__ __forceinline__ void load_n_frags(
        NFrags& f, const unsigned short* __restrict__ HRbf,
        const unsigned short* __restrict__ Wnh,
        int row0, int col0, int k, int lane) {
    f.a0 = load_frag_a(HRbf, KD, row0,      k, lane);
    f.a1 = load_frag_a(HRbf, KD, row0 + 16, k, lane);
    f.b0 = load_frag_b(Wnh, KD, col0,      k, lane);
    f.b1 = load_frag_b(Wnh, KD, col0 + 16, k, lane);
}

static __device__ __forceinline__ void n_mma(const NFrags& f,
        v8f& c00, v8f& c01, v8f& c10, v8f& c11) {
    c00 = wmma_bf16(f.a0, f.b0, c00);
    c10 = wmma_bf16(f.a1, f.b0, c10);
    c01 = wmma_bf16(f.a0, f.b1, c01);
    c11 = wmma_bf16(f.a1, f.b1, c11);
}

__global__ __launch_bounds__(256) void gru_n_out_gemm(
        const unsigned short* __restrict__ HRbf,
        const unsigned short* __restrict__ Wnh,    // [N][K] bf16
        const float* __restrict__ NXin,
        const float* __restrict__ Zin,
        const float* __restrict__ states,
        const float* __restrict__ b_n,
        float* __restrict__ out) {
    const int lane = threadIdx.x & 31;
    const int wave = threadIdx.x >> 5;
    const int wm = wave & 1;              // 0..1
    const int wn = wave >> 1;             // 0..3
    const int row0 = blockIdx.y * 64 + wm * 32;
    const int col0 = blockIdx.x * 128 + wn * 32;

    v8f c00 = {}, c01 = {}, c10 = {}, c11 = {};

    NFrags f0, f1;
    load_n_frags(f0, HRbf, Wnh, row0, col0, 0, lane);

    for (int k = 0; k < KD; k += 64) {
        load_n_frags(f1, HRbf, Wnh, row0, col0, k + 32, lane);
        n_mma(f0, c00, c01, c10, c11);
        if (k + 64 < KD)
            load_n_frags(f0, HRbf, Wnh, row0, col0, k + 64, lane);
        n_mma(f1, c00, c01, c10, c11);
    }

    const int n0 = lane & 15;
    const int hi = lane >> 4;
#pragma unroll
    for (int t = 0; t < 2; ++t) {
        const int col = col0 + t * 16 + n0;
        const float bn = b_n[col];
#pragma unroll
        for (int s = 0; s < 2; ++s) {
            const v8f& acc = t ? (s ? c11 : c01) : (s ? c10 : c00);
#pragma unroll
            for (int v = 0; v < 8; ++v) {
                const int row = row0 + s * 16 + hi * 8 + v;
                const size_t idx = (size_t)row * NU + col;
                float n = tanhf(NXin[idx] + acc[v] + bn);
                float z = Zin[idx];
                float h = states[idx];
                out[idx] = (1.0f - z) * h + z * n;
            }
        }
    }
}

// ---------------- host ----------------

extern "C" void kernel_launch(void* const* d_in, const int* in_sizes, int n_in,
                              void* d_out, int out_size, void* d_ws, size_t ws_size,
                              hipStream_t stream) {
    const float* X   = (const float*)d_in[0];
    const float* H   = (const float*)d_in[1];
    const float* w_r = (const float*)d_in[2];
    const float* b_r = (const float*)d_in[3];
    const float* w_z = (const float*)d_in[4];
    const float* b_z = (const float*)d_in[5];
    const float* w_n = (const float*)d_in[6];
    const float* b_n = (const float*)d_in[7];
    float* out = (float*)d_out;

    char* ws = (char*)d_ws;
    unsigned short* Xbf = (unsigned short*)ws;                 // BB*KD bf16
    unsigned short* Hbf = Xbf + (size_t)BB * KD;               // BB*NU bf16
    unsigned short* Wt  = Hbf + (size_t)BB * NU;               // 6*MSZ bf16
    float* Z   = (float*)(Wt + 6 * MSZ);                       // BB*NU f32
    float* NX  = Z + (size_t)BB * NU;                          // BB*NU f32
    unsigned short* HR = (unsigned short*)(NX + (size_t)BB * NU); // BB*NU bf16

    // 1) convert activations to bf16
    {
        const int n4 = (BB * KD) / 4;                          // 2,097,152
        cvt_f32_bf16<<<n4 / 256, 256, 0, stream>>>(X, Xbf, n4);
        cvt_f32_bf16<<<n4 / 256, 256, 0, stream>>>(H, Hbf, n4);
    }

    // 2) convert + transpose the six 2048x2048 weight sub-matrices
    {
        dim3 g(NU / 32, KD / 32);
        transpose_cvt<<<g, 256, 0, stream>>>(w_r,                   Wt + 0 * MSZ); // Wr_x^T
        transpose_cvt<<<g, 256, 0, stream>>>(w_r + (size_t)KD * NU, Wt + 1 * MSZ); // Wr_h^T
        transpose_cvt<<<g, 256, 0, stream>>>(w_z,                   Wt + 2 * MSZ); // Wz_x^T
        transpose_cvt<<<g, 256, 0, stream>>>(w_z + (size_t)KD * NU, Wt + 3 * MSZ); // Wz_h^T
        transpose_cvt<<<g, 256, 0, stream>>>(w_n,                   Wt + 4 * MSZ); // Wn_x^T
        transpose_cvt<<<g, 256, 0, stream>>>(w_n + (size_t)KD * NU, Wt + 5 * MSZ); // Wn_h^T
    }

    // 3) fused gates GEMM (r, z, Xn) + epilogue
    {
        dim3 g(NU / 64, BB / 64);
        gru_gates_gemm<<<g, 256, 0, stream>>>(Xbf, Hbf, Wt, H, b_r, b_z, Z, NX, HR);
    }

    // 4) final GEMM (h*r)@Wn_h + tanh + blend
    {
        dim3 g(NU / 128, BB / 64);
        gru_n_out_gemm<<<g, 256, 0, stream>>>(HR, Wt + 5 * MSZ, NX, Z, H, b_n, out);
    }
}
